// MultiTypeGraphAttention_29901562314877
// MI455X (gfx1250) — matrix operand order repaired and checked by reading, not compiled, not run on hardware
//
#include <hip/hip_runtime.h>
#include <math.h>

// ---------------------------------------------------------------------------
// Problem constants (from reference)
// ---------------------------------------------------------------------------
#define NSRC   20000
#define NDST   20000
#define NEDGE  640000
#define CDIM   128
#define NHEAD  8
#define DHEAD  16
#define EDIM   6
#define HID1   64
#define HID2   32
#define OC1    256   // 2*C
#define OC2    384   // 3*C

typedef __attribute__((ext_vector_type(16))) _Float16 v16h;
typedef __attribute__((ext_vector_type(8)))  float    v8f;

// ---------------------------------------------------------------------------
// Helpers: ordered-uint encoding of floats for atomic max
// ---------------------------------------------------------------------------
__device__ __forceinline__ unsigned f2ord(float f) {
    unsigned u = __float_as_uint(f);
    return (u & 0x80000000u) ? ~u : (u | 0x80000000u);
}
__device__ __forceinline__ float ord2f(unsigned u) {
    return (u & 0x80000000u) ? __uint_as_float(u & 0x7fffffffu)
                             : __uint_as_float(~u);
}

// ---------------------------------------------------------------------------
// f32 -> f16 convert (activations)
// ---------------------------------------------------------------------------
__global__ void f32_to_f16_kernel(const float* __restrict__ x,
                                  _Float16* __restrict__ y, size_t n) {
    size_t i = (size_t)blockIdx.x * blockDim.x + threadIdx.x;
    if (i < n) y[i] = (_Float16)x[i];
}

// ---------------------------------------------------------------------------
// Pack W [N,K] (row-major f32, torch Linear weight) into WMMA B-fragment-major
// f16 layout: fragment f = (ntile*ktiles + kt), lane l, element j holds
//   W[ ntile*16 + (l&15),  kt*32 + (l&16 ? 16 : 0) + j ]
// so each lane's 16-half B fragment is one contiguous 32-byte v16h, and the
// pack kernel itself reads W contiguously.
// ---------------------------------------------------------------------------
__global__ void pack_b_kernel(const float* __restrict__ W,
                              _Float16* __restrict__ Bp, int N, int K) {
    size_t o = (size_t)blockIdx.x * blockDim.x + threadIdx.x;
    if (o >= (size_t)N * K) return;
    const int ktiles = K >> 5;
    int    j     = (int)(o & 15);
    int    lane  = (int)((o >> 4) & 31);
    size_t f     = o >> 9;                    // 512 elements per fragment
    int    kt    = (int)(f % ktiles);
    int    ntile = (int)(f / ktiles);
    int    n     = ntile * 16 + (lane & 15);
    int    k     = kt * 32 + ((lane & 16) ? 16 : 0) + j;
    Bp[o] = (_Float16)W[(size_t)n * K + k];
}

// Zero the accumulator + softmax stats (must run every launch: harness
// poisons ws once and never restores it between graph replays)
__global__ void init_kernel(float* __restrict__ out_accum, size_t n,
                            float* __restrict__ head_sum,
                            unsigned* __restrict__ head_max) {
    size_t i = (size_t)blockIdx.x * blockDim.x + threadIdx.x;
    if (i < n) out_accum[i] = 0.0f;
    if (i < NHEAD) { head_sum[i] = 0.0f; head_max[i] = 0u; } // 0u == ord(-inf)
}

// ---------------------------------------------------------------------------
// WMMA GEMM:  C[M,N] = act( A[M,K](f16) * B(packed fragments) + bias[N] )
// One wave per 16x64 output strip: A fragment loaded once per K-step and
// reused across 4 v_wmma_f32_16x16x32_f16; B fragments are single v16h loads
// (2x global_load_b128 each) thanks to pack_b_kernel.
// ---------------------------------------------------------------------------
template <bool RELU, bool HALF_OUT>
__global__ __launch_bounds__(256)
void wmma_gemm_kernel(const _Float16* __restrict__ A,
                      const v16h* __restrict__ Bp,
                      const float* __restrict__ bias,
                      float* __restrict__ Cf, _Float16* __restrict__ Ch,
                      int M, int N, int K) {
    const int lane  = threadIdx.x & 31;
    const int wave  = threadIdx.x >> 5;
    const int tileM = blockIdx.x * (blockDim.x >> 5) + wave;
    const int row0  = tileM * 16;
    if (row0 >= M) return;                 // wave-uniform: EXEC stays all-ones
    const int ntile0 = blockIdx.y * 4;     // 4 N-tiles (64 cols) per wave

    const int hi     = (lane & 16) ? 1 : 0;
    const int rowA   = row0 + (lane & 15);
    const int aKoff  = hi ? 16 : 0;        // A-fragment lane-group K offset
    const int ktiles = K >> 5;

    v8f acc[4] = {};
    for (int kt = 0; kt < ktiles; ++kt) {
        v16h a;
        const _Float16* Ap = A + (size_t)rowA * K + kt * 32;
#pragma unroll
        for (int v = 0; v < 8; ++v) {
            // A 16x32 f16: VGPR v holds K = (hi?16:0) + 2*v (matches packed B)
            // -> one contiguous 32-byte run per lane (coalesces to 2x b128)
            int k = aKoff + 2 * v;
            a[2 * v]     = Ap[k];
            a[2 * v + 1] = Ap[k + 1];
        }
#pragma unroll
        for (int nt = 0; nt < 4; ++nt) {
            v16h b = Bp[((size_t)(ntile0 + nt) * ktiles + kt) * 32 + lane];
            acc[nt] = __builtin_amdgcn_wmma_f32_16x16x32_f16(
                /*neg_a=*/false, a, /*neg_b=*/false, b,
                /*c_mod=*/(short)0, acc[nt],
                /*reuse_a=*/false, /*reuse_b=*/false);
        }
    }

#pragma unroll
    for (int nt = 0; nt < 4; ++nt) {
        const int   col = (ntile0 + nt) * 16 + (lane & 15);
        const float bb  = bias[col];
#pragma unroll
        for (int r = 0; r < 8; ++r) {
            // C/D f32 16x16: VGPR r -> row M = r (lanes 0-15) / 8+r (16-31)
            int   row = row0 + r + (hi ? 8 : 0);
            float v   = acc[nt][r] + bb;
            if (RELU) v = fmaxf(v, 0.0f);
            if (HALF_OUT) Ch[(size_t)row * N + col] = (_Float16)v;
            else          Cf[(size_t)row * N + col] = v;
        }
    }
}

// ---------------------------------------------------------------------------
// Pass 1: per-edge attention scores (Q.K per head + gated edge-MLP bias),
// plus global per-head max via ordered-uint atomics (L2-resident).
// ---------------------------------------------------------------------------
__global__ __launch_bounds__(256)
void edge_scores_kernel(const _Float16* __restrict__ Qh,
                        const _Float16* __restrict__ Kh,
                        const float* __restrict__ edge_attr,
                        const int* __restrict__ src, const int* __restrict__ dst,
                        const float* __restrict__ e_w1, const float* __restrict__ e_b1,
                        const float* __restrict__ e_w2, const float* __restrict__ e_b2,
                        const float* __restrict__ e_w3, const float* __restrict__ e_b3,
                        const float* __restrict__ e_ws, const float* __restrict__ e_bs,
                        const float* __restrict__ gate_p,
                        float* __restrict__ scores,
                        unsigned* __restrict__ head_max, int E) {
    __shared__ float w1[HID1 * EDIM], b1[HID1];
    __shared__ float w2[HID2 * HID1], b2[HID2];
    __shared__ float w3[NHEAD * HID2], b3[NHEAD];
    __shared__ float ws_[NHEAD * EDIM], bs_[NHEAD];
    __shared__ unsigned smax[NHEAD];

    for (int i = threadIdx.x; i < HID1 * EDIM; i += blockDim.x) w1[i] = e_w1[i];
    for (int i = threadIdx.x; i < HID1;        i += blockDim.x) b1[i] = e_b1[i];
    for (int i = threadIdx.x; i < HID2 * HID1; i += blockDim.x) w2[i] = e_w2[i];
    for (int i = threadIdx.x; i < HID2;        i += blockDim.x) b2[i] = e_b2[i];
    for (int i = threadIdx.x; i < NHEAD * HID2; i += blockDim.x) w3[i] = e_w3[i];
    for (int i = threadIdx.x; i < NHEAD;       i += blockDim.x) b3[i] = e_b3[i];
    for (int i = threadIdx.x; i < NHEAD * EDIM; i += blockDim.x) ws_[i] = e_ws[i];
    for (int i = threadIdx.x; i < NHEAD;       i += blockDim.x) bs_[i] = e_bs[i];
    if (threadIdx.x < NHEAD) smax[threadIdx.x] = 0u;
    __syncthreads();

    int e = blockIdx.x * blockDim.x + threadIdx.x;
    if (e < E) {
        const int s = src[e], d = dst[e];
        const _Float16* q = Qh + (size_t)d * CDIM;
        const _Float16* k = Kh + (size_t)s * CDIM;

        float sc[NHEAD];
#pragma unroll
        for (int h = 0; h < NHEAD; ++h) {
            float acc = 0.0f;
#pragma unroll
            for (int i = 0; i < DHEAD; ++i)
                acc += (float)q[h * DHEAD + i] * (float)k[h * DHEAD + i];
            sc[h] = acc * 0.25f;                       // 1/sqrt(16)
        }

        float ea[EDIM];
#pragma unroll
        for (int i = 0; i < EDIM; ++i) ea[i] = edge_attr[(size_t)e * EDIM + i];

        float h1[HID1];
        for (int j = 0; j < HID1; ++j) {
            float a = b1[j];
#pragma unroll
            for (int i = 0; i < EDIM; ++i) a += ea[i] * w1[j * EDIM + i];
            h1[j] = fmaxf(a, 0.0f);
        }
        float h2[HID2];
        for (int j = 0; j < HID2; ++j) {
            float a = b2[j];
            for (int i = 0; i < HID1; ++i) a += h1[i] * w2[j * HID1 + i];
            h2[j] = fmaxf(a, 0.0f);
        }
        const float gate = 1.0f / (1.0f + __expf(-gate_p[0]));
#pragma unroll
        for (int h = 0; h < NHEAD; ++h) {
            float m = b3[h];
            for (int i = 0; i < HID2; ++i) m += h2[i] * w3[h * HID2 + i];
            float sh = bs_[h];
#pragma unroll
            for (int i = 0; i < EDIM; ++i) sh += ea[i] * ws_[h * EDIM + i];
            float v = sc[h] + gate * m + (1.0f - gate) * sh;
            scores[(size_t)e * NHEAD + h] = v;
            atomicMax(&smax[h], f2ord(v));
        }
    }
    __syncthreads();
    if (threadIdx.x < NHEAD)
        atomicMax(&head_max[threadIdx.x], smax[threadIdx.x]);
}

// ---------------------------------------------------------------------------
// Pass 2: exp(score - max) in place, per-head global sums
// ---------------------------------------------------------------------------
__global__ __launch_bounds__(256)
void edge_expsum_kernel(float* __restrict__ scores,
                        const unsigned* __restrict__ head_max,
                        float* __restrict__ head_sum, int E) {
    __shared__ float ssum[NHEAD];
    __shared__ float smax[NHEAD];
    if (threadIdx.x < NHEAD) {
        ssum[threadIdx.x] = 0.0f;
        smax[threadIdx.x] = ord2f(head_max[threadIdx.x]);
    }
    __syncthreads();
    int e = blockIdx.x * blockDim.x + threadIdx.x;
    if (e < E) {
#pragma unroll
        for (int h = 0; h < NHEAD; ++h) {
            float ex = __expf(scores[(size_t)e * NHEAD + h] - smax[h]);
            scores[(size_t)e * NHEAD + h] = ex;
            atomicAdd(&ssum[h], ex);
        }
    }
    __syncthreads();
    if (threadIdx.x < NHEAD)
        atomicAdd(&head_sum[threadIdx.x], ssum[threadIdx.x]);
}

// ---------------------------------------------------------------------------
// Pass 3: weighted message scatter-add (global_atomic_add_f32, L2-resident)
// block = one edge, thread = one channel
// ---------------------------------------------------------------------------
__global__ __launch_bounds__(128)
void scatter_kernel(const float* __restrict__ scores,
                    const float* __restrict__ head_sum,
                    const _Float16* __restrict__ Vh,
                    const int* __restrict__ src, const int* __restrict__ dst,
                    float* __restrict__ out_accum) {
    const int e = blockIdx.x;
    const int c = threadIdx.x;
    const int h = c >> 4;                  // D = 16 channels per head
    const float w = scores[(size_t)e * NHEAD + h] / head_sum[h];
    const float v = (float)Vh[(size_t)src[e] * CDIM + c];
    atomicAdd(out_accum + (size_t)dst[e] * CDIM + c, v * w);
}

// ---------------------------------------------------------------------------
// Residual + pos-encoding + LayerNorm (eps = 1e-5), one row per block
// ---------------------------------------------------------------------------
__global__ __launch_bounds__(128)
void ln_kernel(const float* __restrict__ o, const float* __restrict__ x_dst,
               const float* __restrict__ pos, const float* __restrict__ g,
               const float* __restrict__ b, float* __restrict__ out) {
    __shared__ float red[CDIM];
    const int row = blockIdx.x;
    const int c   = threadIdx.x;
    float z = o[(size_t)row * CDIM + c] + x_dst[(size_t)row * CDIM + c] + pos[c];
    red[c] = z;
    __syncthreads();
    for (int s = CDIM / 2; s > 0; s >>= 1) {
        if (c < s) red[c] += red[c + s];
        __syncthreads();
    }
    float mu = red[0] / (float)CDIM;
    __syncthreads();
    float dz = z - mu;
    red[c] = dz * dz;
    __syncthreads();
    for (int s = CDIM / 2; s > 0; s >>= 1) {
        if (c < s) red[c] += red[c + s];
        __syncthreads();
    }
    float var = red[0] / (float)CDIM;
    out[(size_t)row * CDIM + c] = dz * rsqrtf(var + 1e-5f) * g[c] + b[c];
}

// ---------------------------------------------------------------------------
// Host-side orchestration
// ---------------------------------------------------------------------------
static inline size_t align_up(size_t x) { return (x + 255) & ~(size_t)255; }

extern "C" void kernel_launch(void* const* d_in, const int* in_sizes, int n_in,
                              void* d_out, int out_size, void* d_ws, size_t ws_size,
                              hipStream_t stream) {
    (void)in_sizes; (void)n_in; (void)out_size; (void)ws_size;

    const float* x_src  = (const float*)d_in[0];
    const float* x_dst  = (const float*)d_in[1];
    const float* e_attr = (const float*)d_in[2];
    const float* Wq = (const float*)d_in[3];  const float* bq = (const float*)d_in[4];
    const float* Wk = (const float*)d_in[5];  const float* bk = (const float*)d_in[6];
    const float* Wv = (const float*)d_in[7];  const float* bv = (const float*)d_in[8];
    const float* e_w1 = (const float*)d_in[9];  const float* e_b1 = (const float*)d_in[10];
    const float* e_w2 = (const float*)d_in[11]; const float* e_b2 = (const float*)d_in[12];
    const float* e_w3 = (const float*)d_in[13]; const float* e_b3 = (const float*)d_in[14];
    const float* e_ws = (const float*)d_in[15]; const float* e_bs = (const float*)d_in[16];
    const float* gate = (const float*)d_in[17];
    const float* o_w1 = (const float*)d_in[18]; const float* o_b1 = (const float*)d_in[19];
    const float* o_w2 = (const float*)d_in[20]; const float* o_b2 = (const float*)d_in[21];
    const float* o_w3 = (const float*)d_in[22]; const float* o_b3 = (const float*)d_in[23];
    const float* pos  = (const float*)d_in[24];
    const float* ln_g = (const float*)d_in[25]; const float* ln_b = (const float*)d_in[26];
    const int*   eidx = (const int*)d_in[27];
    const int*   src  = eidx;          // edge_index[0]
    const int*   dst  = eidx + NEDGE;  // edge_index[1]

    // --- workspace carve-up ---
    char*  base = (char*)d_ws;
    size_t off  = 0;
    auto carve = [&](size_t bytes) { char* p = base + off; off = align_up(off + bytes); return p; };

    _Float16* xsrc_h   = (_Float16*)carve((size_t)NSRC * CDIM * 2);
    _Float16* xdst_h   = (_Float16*)carve((size_t)NDST * CDIM * 2);
    _Float16* WqP      = (_Float16*)carve((size_t)CDIM * CDIM * 2);
    _Float16* WkP      = (_Float16*)carve((size_t)CDIM * CDIM * 2);
    _Float16* WvP      = (_Float16*)carve((size_t)CDIM * CDIM * 2);
    _Float16* Qh       = (_Float16*)carve((size_t)NDST * CDIM * 2);
    _Float16* Kh       = (_Float16*)carve((size_t)NSRC * CDIM * 2);
    _Float16* Vh       = (_Float16*)carve((size_t)NSRC * CDIM * 2);
    float*    scores   = (float*)   carve((size_t)NEDGE * NHEAD * 4);
    unsigned* head_max = (unsigned*)carve(NHEAD * 4);
    float*    head_sum = (float*)   carve(NHEAD * 4);
    float*    out_acc  = (float*)   carve((size_t)NDST * CDIM * 4);
    _Float16* act0     = (_Float16*)carve((size_t)NDST * CDIM * 2);
    _Float16* Wo1P     = (_Float16*)carve((size_t)OC1 * CDIM * 2);
    _Float16* Wo2P     = (_Float16*)carve((size_t)OC2 * OC1 * 2);
    _Float16* Wo3P     = (_Float16*)carve((size_t)CDIM * OC2 * 2);
    _Float16* act1     = (_Float16*)carve((size_t)NDST * OC1 * 2);
    _Float16* act2     = (_Float16*)carve((size_t)NDST * OC2 * 2);
    float*    ofinal   = (float*)   carve((size_t)NDST * CDIM * 4);

    // --- conversions + weight fragment packing ---
    {
        size_t n = (size_t)NSRC * CDIM;
        f32_to_f16_kernel<<<(n + 255) / 256, 256, 0, stream>>>(x_src, xsrc_h, n);
        n = (size_t)NDST * CDIM;
        f32_to_f16_kernel<<<(n + 255) / 256, 256, 0, stream>>>(x_dst, xdst_h, n);
    }
    pack_b_kernel<<<(CDIM * CDIM + 255) / 256, 256, 0, stream>>>(Wq, WqP, CDIM, CDIM);
    pack_b_kernel<<<(CDIM * CDIM + 255) / 256, 256, 0, stream>>>(Wk, WkP, CDIM, CDIM);
    pack_b_kernel<<<(CDIM * CDIM + 255) / 256, 256, 0, stream>>>(Wv, WvP, CDIM, CDIM);
    pack_b_kernel<<<(OC1 * CDIM + 255) / 256, 256, 0, stream>>>(o_w1, Wo1P, OC1, CDIM);
    pack_b_kernel<<<(OC2 * OC1 + 255) / 256, 256, 0, stream>>>(o_w2, Wo2P, OC2, OC1);
    pack_b_kernel<<<(CDIM * OC2 + 255) / 256, 256, 0, stream>>>(o_w3, Wo3P, CDIM, OC2);

    // --- zero accumulators & stats (every call) ---
    {
        size_t n = (size_t)NDST * CDIM;
        init_kernel<<<(n + 255) / 256, 256, 0, stream>>>(out_acc, n, head_sum, head_max);
    }

    // --- QKV projections (WMMA) ---
    const int mtiles = NDST / 16;                 // 1250
    const int gx     = (mtiles + 7) / 8;          // 157 blocks of 8 waves
    dim3 gQKV(gx, CDIM / 64);                     // 157 x 2
    wmma_gemm_kernel<false, true><<<gQKV, 256, 0, stream>>>(
        xdst_h, (const v16h*)WqP, bq, nullptr, Qh, NDST, CDIM, CDIM);
    wmma_gemm_kernel<false, true><<<gQKV, 256, 0, stream>>>(
        xsrc_h, (const v16h*)WkP, bk, nullptr, Kh, NSRC, CDIM, CDIM);
    wmma_gemm_kernel<false, true><<<gQKV, 256, 0, stream>>>(
        xsrc_h, (const v16h*)WvP, bv, nullptr, Vh, NSRC, CDIM, CDIM);

    // --- edge scores + global softmax + scatter ---
    const int eb = (NEDGE + 255) / 256;           // 2500
    edge_scores_kernel<<<eb, 256, 0, stream>>>(Qh, Kh, e_attr, src, dst,
                                               e_w1, e_b1, e_w2, e_b2, e_w3, e_b3,
                                               e_ws, e_bs, gate, scores, head_max, NEDGE);
    edge_expsum_kernel<<<eb, 256, 0, stream>>>(scores, head_max, head_sum, NEDGE);
    scatter_kernel<<<NEDGE, CDIM, 0, stream>>>(scores, head_sum, Vh, src, dst, out_acc);

    // --- output MLP (WMMA chain) ---
    {
        size_t n = (size_t)NDST * CDIM;
        f32_to_f16_kernel<<<(n + 255) / 256, 256, 0, stream>>>(out_acc, act0, n);
    }
    dim3 g1(gx, OC1 / 64);                        // 157 x 4
    dim3 g2(gx, OC2 / 64);                        // 157 x 6
    dim3 g3(gx, CDIM / 64);                       // 157 x 2
    wmma_gemm_kernel<true,  true ><<<g1, 256, 0, stream>>>(
        act0, (const v16h*)Wo1P, o_b1, nullptr, act1, NDST, OC1, CDIM);
    wmma_gemm_kernel<true,  true ><<<g2, 256, 0, stream>>>(
        act1, (const v16h*)Wo2P, o_b2, nullptr, act2, NDST, OC2, OC1);
    wmma_gemm_kernel<false, false><<<g3, 256, 0, stream>>>(
        act2, (const v16h*)Wo3P, o_b3, ofinal, nullptr, NDST, CDIM, OC2);

    // --- residual + posenc + LayerNorm ---
    ln_kernel<<<NDST, CDIM, 0, stream>>>(ofinal, x_dst, pos, ln_g, ln_b, (float*)d_out);
}
